// GrokBaselineModel_26792005992840
// MI455X (gfx1250) — matrix operand order repaired and checked by reading, not compiled, not run on hardware
//
#include <hip/hip_runtime.h>
#include <math.h>

typedef __bf16 bf16;
typedef __attribute__((ext_vector_type(16))) __bf16 v16bf;
typedef __attribute__((ext_vector_type(8)))  __bf16 v8bf;
typedef __attribute__((ext_vector_type(8)))  float  v8f;

#define BB      2
#define SS      2048
#define HH      1024
#define NHEAD   16
#define NLAYER  4
#define DD      64
#define FF      2736
#define PREFIX  1536
#define EPSF    1e-6f
#define CAPF    30.0f
#define NEGF    -10000.0f

union FragBF { v16bf v; v8bf h[2]; };

static __device__ __forceinline__ v8f wmma_bf16(v16bf a, v16bf b, v8f c) {
  return __builtin_amdgcn_wmma_f32_16x16x32_bf16(false, a, false, b, (short)0, c, false, false);
}

// ---------------------------------------------------------------------------
// CDNA5 async global->LDS copy (ASYNCcnt path): 16B per lane, no VGPR bounce.
// ---------------------------------------------------------------------------
static __device__ __forceinline__ void async_ld16(unsigned ldsOff, const void* g) {
  unsigned long long ga = (unsigned long long)(uintptr_t)g;
  asm volatile("global_load_async_to_lds_b128 %0, %1, off"
               :: "v"(ldsOff), "v"(ga) : "memory");
}
static __device__ __forceinline__ void wait_async0() {
  asm volatile("s_wait_asynccnt 0" ::: "memory");
}
static __device__ __forceinline__ void wait_ds0() {
  asm volatile("s_wait_dscnt 0" ::: "memory");
}
static __device__ __forceinline__ unsigned lds_off(const void* p) {
  return (unsigned)(uintptr_t)p;  // flat shared addr: low 32 bits = LDS offset
}

// ---------------------------------------------------------------------------
// Weight cast + transpose: W fp32 [N][K] -> Wt bf16 [K][N]
// ---------------------------------------------------------------------------
__global__ void castT_kernel(const float* __restrict__ W, bf16* __restrict__ Wt,
                             int N, int K) {
  size_t idx = (size_t)blockIdx.x * blockDim.x + threadIdx.x;
  size_t total = (size_t)N * K;
  if (idx >= total) return;
  int n = (int)(idx / K);
  int k = (int)(idx % K);
  Wt[(size_t)k * N + n] = (bf16)W[idx];
}

// ---------------------------------------------------------------------------
// RMSNorm (fp32 in) -> bf16 out, one block (256 thr) per row of H=1024
// ---------------------------------------------------------------------------
__global__ __launch_bounds__(256) void rmsnorm_bf16_kernel(
    const float* __restrict__ X, const float* __restrict__ g, bf16* __restrict__ Y) {
  __shared__ float red[256];
  const int row = blockIdx.x, t = threadIdx.x;
  float v[4], ss = 0.f;
  for (int e = 0; e < 4; e++) {
    int c = t + e * 256;
    v[e] = X[(size_t)row * HH + c];
    ss += v[e] * v[e];
  }
  red[t] = ss; __syncthreads();
  for (int s = 128; s > 0; s >>= 1) { if (t < s) red[t] += red[t + s]; __syncthreads(); }
  float rms = rsqrtf(red[0] / (float)HH + EPSF);
  for (int e = 0; e < 4; e++) {
    int c = t + e * 256;
    Y[(size_t)row * HH + c] = (bf16)(v[e] * rms * g[c]);
  }
}

// ---------------------------------------------------------------------------
// Residual + post RMSNorm: h += rmsnorm(preZero? mask(y):y, g); postZero masks h
// ---------------------------------------------------------------------------
__global__ __launch_bounds__(256) void resid_post_kernel(
    float* __restrict__ h, const float* __restrict__ y, const float* __restrict__ g,
    const unsigned char* __restrict__ pad, int preZero, int postZero) {
  __shared__ float red[256];
  const int row = blockIdx.x, t = threadIdx.x;
  const bool p = pad[row] != 0;
  float v[4], ss = 0.f;
  for (int e = 0; e < 4; e++) {
    int c = t + e * 256;
    float x = y[(size_t)row * HH + c];
    if (preZero && p) x = 0.f;
    v[e] = x; ss += x * x;
  }
  red[t] = ss; __syncthreads();
  for (int s = 128; s > 0; s >>= 1) { if (t < s) red[t] += red[t + s]; __syncthreads(); }
  float rms = rsqrtf(red[0] / (float)HH + EPSF);
  for (int e = 0; e < 4; e++) {
    int c = t + e * 256;
    size_t idx = (size_t)row * HH + c;
    float nh = h[idx] + v[e] * rms * g[c];
    if (postZero && p) nh = 0.f;
    h[idx] = nh;
  }
}

// ---------------------------------------------------------------------------
// Final RMSNorm -> fp32 out
// ---------------------------------------------------------------------------
__global__ __launch_bounds__(256) void final_rms_kernel(
    const float* __restrict__ X, const float* __restrict__ g, float* __restrict__ Y) {
  __shared__ float red[256];
  const int row = blockIdx.x, t = threadIdx.x;
  float v[4], ss = 0.f;
  for (int e = 0; e < 4; e++) {
    int c = t + e * 256;
    v[e] = X[(size_t)row * HH + c];
    ss += v[e] * v[e];
  }
  red[t] = ss; __syncthreads();
  for (int s = 128; s > 0; s >>= 1) { if (t < s) red[t] += red[t + s]; __syncthreads(); }
  float rms = rsqrtf(red[0] / (float)HH + EPSF);
  for (int e = 0; e < 4; e++) {
    int c = t + e * 256;
    Y[(size_t)row * HH + c] = v[e] * rms * g[c];
  }
}

// ---------------------------------------------------------------------------
// RoPE. q keeps [B*S][H] layout (bf16); k written transposed [b*NH+h][D][S].
// ---------------------------------------------------------------------------
static __device__ __forceinline__ float rope_val(const float* __restrict__ in,
                                                 size_t idx, int d, int s) {
  float x = in[idx];
  float xo = (d < 32) ? -in[idx + 32] : in[idx - 32];
  float inv = powf(10000.f, -(float)(d & ~1) / 64.f);
  float ph = (float)s * inv;
  return x * cosf(ph) + xo * sinf(ph);
}

__global__ void rope_q_kernel(const float* __restrict__ in, bf16* __restrict__ out) {
  size_t idx = (size_t)blockIdx.x * blockDim.x + threadIdx.x;
  if (idx >= (size_t)BB * SS * HH) return;
  int d = (int)(idx & 63);
  int s = (int)((idx >> 10) & (SS - 1));
  out[idx] = (bf16)rope_val(in, idx, d, s);
}

__global__ void rope_kT_kernel(const float* __restrict__ in, bf16* __restrict__ out) {
  size_t idx = (size_t)blockIdx.x * blockDim.x + threadIdx.x;
  if (idx >= (size_t)BB * SS * HH) return;
  int col = (int)(idx & (HH - 1));
  int d = col & 63, h = col >> 6;
  int s = (int)((idx >> 10) & (SS - 1));
  int b = (int)(idx >> 21);
  float r = rope_val(in, idx, d, s);
  out[((size_t)((b * NHEAD + h) * DD + d)) * SS + s] = (bf16)r;
}

__global__ void cast_bf16_kernel(const float* __restrict__ in, bf16* __restrict__ out, size_t n) {
  size_t idx = (size_t)blockIdx.x * blockDim.x + threadIdx.x;
  if (idx < n) out[idx] = (bf16)in[idx];
}

__global__ void gelu_gate_kernel(const float* __restrict__ g, const float* __restrict__ v,
                                 bf16* __restrict__ out, size_t n) {
  size_t idx = (size_t)blockIdx.x * blockDim.x + threadIdx.x;
  if (idx >= n) return;
  float x = g[idx];
  float ge = 0.5f * x * (1.f + erff(x * 0.70710678118654752f));
  out[idx] = (bf16)(ge * v[idx]);
}

// ---------------------------------------------------------------------------
// WMMA GEMM: Y[M][N] = X[M][K] * Wt[K][N]  (X * W^T with W pre-transposed).
// 256 threads = 8 waves, block tile 128x128, K step 32, wave tile 64x32.
// Double-buffered LDS; interior tiles staged via async global->LDS DMA.
// ---------------------------------------------------------------------------
#define GBM 128
#define GBN 128
#define GBK 32

static __device__ __forceinline__ void gemm_stage(
    const bf16* __restrict__ X, const bf16* __restrict__ Wt,
    bf16 (*sXb)[GBK], bf16 (*sWb)[GBN],
    int M, int N, int K, int m0, int n0, int k0, int t) {
  // --- X tile [128][32]: thread -> 16 bf16 (32B) ---
  {
    const int row = t >> 1, col = (t & 1) * 16;
    const int gm = m0 + row, gk = k0 + col;
    bf16* dst = &sXb[row][col];
    if ((m0 + GBM <= M) && (k0 + GBK <= K)) {
      const bf16* src = X + (size_t)gm * K + gk;
      unsigned d = lds_off(dst);
      async_ld16(d, src);
      async_ld16(d + 16, src + 8);
    } else {
      for (int e = 0; e < 16; e++)
        dst[e] = (gm < M && gk + e < K) ? X[(size_t)gm * K + gk + e] : (bf16)0.0f;
    }
  }
  // --- Wt tile [32][128]: thread -> 16 bf16 ---
  {
    const int row = t >> 3, col = (t & 7) * 16;
    const int gk = k0 + row, gn = n0 + col;
    bf16* dst = &sWb[row][col];
    if ((k0 + GBK <= K) && (n0 + GBN <= N)) {
      const bf16* src = Wt + (size_t)gk * N + gn;
      unsigned d = lds_off(dst);
      async_ld16(d, src);
      async_ld16(d + 16, src + 8);
    } else {
      for (int e = 0; e < 16; e++)
        dst[e] = (gk < K && gn + e < N) ? Wt[(size_t)gk * N + gn + e] : (bf16)0.0f;
    }
  }
}

__global__ __launch_bounds__(256) void gemm_xwt_kernel(
    const bf16* __restrict__ X, const bf16* __restrict__ Wt, float* __restrict__ Y,
    int M, int N, int K) {
  __shared__ __align__(16) bf16 sX[2][GBM][GBK];
  __shared__ __align__(16) bf16 sW[2][GBK][GBN];

  const int m0 = blockIdx.y * GBM;
  const int n0 = blockIdx.x * GBN;
  const int t = threadIdx.x;
  const int lane = t & 31;
  const int w = t >> 5;
  const int wm = (w >> 2) * 64;   // 0 or 64
  const int wn = (w & 3) * 32;    // 0..96
  const int half = lane >> 4;
  const int l16 = lane & 15;

  v8f acc[4][2];
  for (int i = 0; i < 4; i++)
    for (int j = 0; j < 2; j++)
      for (int r = 0; r < 8; r++) acc[i][j][r] = 0.f;

  const int nsteps = (K + GBK - 1) / GBK;
  gemm_stage(X, Wt, sX[0], sW[0], M, N, K, m0, n0, 0, t);   // prologue DMA

  for (int s = 0; s < nsteps; s++) {
    const int cur = s & 1;
    wait_async0();          // my DMA for buf[cur] done
    wait_ds0();             // my edge-path stores / prev reads done
    __syncthreads();        // buf[cur] visible everywhere; buf[!cur] free
    if (s + 1 < nsteps)     // overlap next-tile DMA with WMMA below
      gemm_stage(X, Wt, sX[cur ^ 1], sW[cur ^ 1], M, N, K, m0, n0, (s + 1) * GBK, t);

    bf16 (*sXc)[GBK] = sX[cur];
    bf16 (*sWc)[GBN] = sW[cur];

    // A fragments: lane l16 holds matrix row, K split per 16-lane half
    FragBF a[4];
    for (int i = 0; i < 4; i++) {
      const bf16* p = &sXc[wm + i * 16 + l16][half * 8];
      a[i].h[0] = *(const v8bf*)p;
      a[i].h[1] = *(const v8bf*)(p + 16);
    }
    // B fragments: lane holds N column l16, K in table order
    FragBF b[2];
    for (int j = 0; j < 2; j++)
      for (int e = 0; e < 16; e++)
        b[j].v[e] = sWc[(e < 8) ? (half * 8 + e) : (8 + half * 8 + e)][wn + j * 16 + l16];
    for (int i = 0; i < 4; i++)
      for (int j = 0; j < 2; j++)
        acc[i][j] = wmma_bf16(a[i].v, b[j].v, acc[i][j]);
  }

  // store (C layout: VGPR r -> M = half*8 + r, N = l16)
  for (int i = 0; i < 4; i++)
    for (int j = 0; j < 2; j++) {
      const int gn = n0 + wn + j * 16 + l16;
      if (gn >= N) continue;
      for (int r = 0; r < 8; r++) {
        const int gm = m0 + wm + i * 16 + half * 8 + r;
        if (gm < M) Y[(size_t)gm * N + gn] = acc[i][j][r];
      }
    }
}

// ---------------------------------------------------------------------------
// Flash attention with tanh cap + analytic mask.
// Grid: (B*NH, S/64). 128 threads = 4 waves, each wave owns 16 query rows.
// K/V tiles staged by async global->LDS DMA.
// ---------------------------------------------------------------------------
static __device__ __forceinline__ bool blkNeeded(int imin, int imax, int j0) {
  if (imax < PREFIX) return j0 <= imax;
  return (j0 < PREFIX) || (j0 <= imax && (j0 + 127) >= imin);
}

__global__ __launch_bounds__(128) void attn_kernel(
    const bf16* __restrict__ Q,            // [B*S][H], roped
    const bf16* __restrict__ Kt,           // [(b*NH+h)*D][S], roped, transposed
    const bf16* __restrict__ V,            // [B*S][H]
    const unsigned char* __restrict__ pad, // [B][S]
    bf16* __restrict__ O) {                // [B*S][H]
  __shared__ __align__(16) bf16 sKt[DD][128];
  __shared__ __align__(16) bf16 sV[128][DD];
  __shared__ __align__(16) bf16 sP[4][16][32];

  const int bh = blockIdx.x;
  const int b = bh >> 4, h = bh & 15;
  const int tile = blockIdx.y;
  const int t = threadIdx.x;
  const int lane = t & 31;
  const int w = t >> 5;
  const int half = lane >> 4;
  const int l16 = lane & 15;
  const int qbase = tile * 64 + w * 16;
  const int iMin = qbase, iMax = qbase + 15;
  const int iMinWG = tile * 64, iMaxWG = tile * 64 + 63;
  const float scale = 0.125f; // D^-0.5

  // Q A-fragments for both 32-wide K chunks of D=64 (reused for all keys)
  FragBF aq[2];
  {
    const bf16* qrow = Q + ((size_t)(b * SS + qbase + l16)) * HH + h * DD;
    for (int c = 0; c < 2; c++) {
      aq[c].h[0] = *(const v8bf*)(qrow + c * 32 + half * 8);
      aq[c].h[1] = *(const v8bf*)(qrow + c * 32 + 16 + half * 8);
    }
  }

  v8f accO[4];
  float mrun[8], lrun[8];
  for (int f = 0; f < 4; f++)
    for (int r = 0; r < 8; r++) accO[f][r] = 0.f;
  for (int r = 0; r < 8; r++) { mrun[r] = -3.0e38f; lrun[r] = 0.f; }

  for (int j0 = 0; j0 < SS; j0 += 128) {
    __syncthreads();                       // previous block fully consumed
    if (blkNeeded(iMinWG, iMaxWG, j0)) {   // block-uniform
      { // K^T block: rows = D, cols = 128 keys (contiguous in Kt)
        const int row = t >> 1, col = (t & 1) * 64;
        const bf16* src = Kt + ((size_t)(bh * DD + row)) * SS + j0 + col;
        unsigned dst = lds_off(&sKt[row][col]);
        for (int e = 0; e < 8; e++) async_ld16(dst + e * 16, src + e * 8);
      }
      { // V block row-major
        const bf16* src = V + ((size_t)(b * SS + j0 + t)) * HH + h * DD;
        unsigned dst = lds_off(&sV[t][0]);
        for (int e = 0; e < 8; e++) async_ld16(dst + e * 16, src + e * 8);
      }
    }
    wait_async0();
    __syncthreads();
    if (!blkNeeded(iMin, iMax, j0)) continue; // wave-level skip (barriers uniform)

    for (int p = 0; p < 4; p++) {             // 32 keys per iter (two 16-col tiles)
      const int jb = j0 + p * 32;
      v8f cl0, cl1;
      for (int r = 0; r < 8; r++) { cl0[r] = 0.f; cl1[r] = 0.f; }
      for (int c = 0; c < 2; c++) {
        FragBF bk0, bk1;
        for (int e = 0; e < 16; e++) {
          const int kr = c * 32 + ((e < 8) ? (half * 8 + e) : (8 + half * 8 + e));
          bk0.v[e] = sKt[kr][p * 32 + l16];
          bk1.v[e] = sKt[kr][p * 32 + 16 + l16];
        }
        cl0 = wmma_bf16(aq[c].v, bk0.v, cl0);
        cl1 = wmma_bf16(aq[c].v, bk1.v, cl1);
      }
      const int jc0 = jb + l16;
      const int jc1 = jb + 16 + l16;
      const bool pk0 = pad[b * SS + jc0] != 0;
      const bool pk1 = pad[b * SS + jc1] != 0;
      float pv0[8], pv1[8];
      for (int r = 0; r < 8; r++) {
        const int i = qbase + half * 8 + r;
        float x0 = cl0[r] * scale; x0 = CAPF * tanhf(x0 * (1.f / CAPF));
        float x1 = cl1[r] * scale; x1 = CAPF * tanhf(x1 * (1.f / CAPF));
        const bool a0 = (i < PREFIX) ? (jc0 <= i) : (jc0 < PREFIX || jc0 == i);
        const bool a1 = (i < PREFIX) ? (jc1 <= i) : (jc1 < PREFIX || jc1 == i);
        x0 = (a0 && !pk0) ? x0 : NEGF;
        x1 = (a1 && !pk1) ? x1 : NEGF;
        float rmax = fmaxf(x0, x1);
        for (int off = 8; off >= 1; off >>= 1) rmax = fmaxf(rmax, __shfl_xor(rmax, off, 16));
        const float mnew = fmaxf(mrun[r], rmax);
        const float alpha = __expf(mrun[r] - mnew);
        const float e0 = __expf(x0 - mnew), e1 = __expf(x1 - mnew);
        float rs = e0 + e1;
        for (int off = 8; off >= 1; off >>= 1) rs += __shfl_xor(rs, off, 16);
        lrun[r] = lrun[r] * alpha + rs;
        mrun[r] = mnew;
        for (int f = 0; f < 4; f++) accO[f][r] *= alpha;
        pv0[r] = e0; pv1[r] = e1;
      }
      // C-layout -> A-layout through per-wave LDS (DS ops are in-order per wave)
      for (int r = 0; r < 8; r++) {
        const int m = half * 8 + r;
        sP[w][m][l16] = (bf16)pv0[r];
        sP[w][m][16 + l16] = (bf16)pv1[r];
      }
      wait_ds0();
      FragBF pa;
      pa.h[0] = *(const v8bf*)&sP[w][l16][half * 8];
      pa.h[1] = *(const v8bf*)&sP[w][l16][16 + half * 8];
      for (int f = 0; f < 4; f++) {
        FragBF bv;
        for (int e = 0; e < 16; e++)
          bv.v[e] = sV[p * 32 + ((e < 8) ? (half * 8 + e) : (8 + half * 8 + e))][f * 16 + l16];
        accO[f] = wmma_bf16(pa.v, bv.v, accO[f]);
      }
    }
  }

  for (int f = 0; f < 4; f++)
    for (int r = 0; r < 8; r++) {
      const int i = qbase + half * 8 + r;
      const float o = accO[f][r] / lrun[r];
      O[((size_t)(b * SS + i)) * HH + h * DD + f * 16 + l16] = (bf16)o;
    }
}

// ---------------------------------------------------------------------------
// Host orchestration
// ---------------------------------------------------------------------------
extern "C" void kernel_launch(void* const* d_in, const int* in_sizes, int n_in,
                              void* d_out, int out_size, void* d_ws, size_t ws_size,
                              hipStream_t stream) {
  (void)in_sizes; (void)n_in; (void)out_size; (void)ws_size;

  const float* hid = (const float*)d_in[0];
  const unsigned char* pad = (const unsigned char*)d_in[1];
  // d_in[2] = allowed_attention (mask is computed analytically in-kernel)
  const float* Wq  = (const float*)d_in[3];
  const float* Wk  = (const float*)d_in[4];
  const float* Wv  = (const float*)d_in[5];
  const float* Wo  = (const float*)d_in[6];
  const float* Wg  = (const float*)d_in[7];
  const float* Wvf = (const float*)d_in[8];
  const float* Wof = (const float*)d_in[9];
  const float* nPreA  = (const float*)d_in[10];
  const float* nPostA = (const float*)d_in[11];
  const float* nPreF  = (const float*)d_in[12];
  const float* nPostF = (const float*)d_in[13];
  const float* nFinal = (const float*)d_in[14];

  const size_t M = (size_t)BB * SS;          // 4096
  const size_t HxH = (size_t)HH * HH;        // 1048576
  const size_t FxH = (size_t)FF * HH;        // 2801664

  // workspace layout
  char* ws = (char*)d_ws;
  size_t off = 0;
  auto alloc = [&](size_t bytes) { size_t o = off; off = (off + bytes + 255) & ~(size_t)255; return o; };
  const size_t oBW    = alloc((4 * HxH + 3 * FxH) * sizeof(bf16)); // per-layer bf16 weights
  const size_t oH     = alloc(M * HH * sizeof(float));
  const size_t oXB    = alloc(M * HH * sizeof(bf16));
  const size_t oBig1  = alloc(M * FF * sizeof(float));
  const size_t oBig2  = alloc(M * FF * sizeof(float));
  const size_t oSml   = alloc(M * HH * sizeof(float));
  const size_t oQB    = alloc(M * HH * sizeof(bf16));
  const size_t oKT    = alloc(M * HH * sizeof(bf16));
  const size_t oVB    = alloc(M * HH * sizeof(bf16));
  const size_t oATT   = alloc(M * HH * sizeof(bf16));
  const size_t oGATE  = alloc(M * FF * sizeof(bf16));

  bf16* bw   = (bf16*)(ws + oBW);
  float* hbuf = (float*)(ws + oH);
  bf16* xb   = (bf16*)(ws + oXB);
  float* big1 = (float*)(ws + oBig1);
  float* big2 = (float*)(ws + oBig2);
  float* sml  = (float*)(ws + oSml);
  bf16* qb   = (bf16*)(ws + oQB);
  bf16* kt   = (bf16*)(ws + oKT);
  bf16* vb   = (bf16*)(ws + oVB);
  bf16* attb = (bf16*)(ws + oATT);
  bf16* gate = (bf16*)(ws + oGATE);

  bf16* WqT  = bw;
  bf16* WkT  = bw + HxH;
  bf16* WvT  = bw + 2 * HxH;
  bf16* WoT  = bw + 3 * HxH;
  bf16* WgT  = bw + 4 * HxH;
  bf16* WvfT = bw + 4 * HxH + FxH;
  bf16* WofT = bw + 4 * HxH + 2 * FxH;

  auto castT = [&](const float* W, bf16* Wt, int N, int K) {
    size_t n = (size_t)N * K;
    castT_kernel<<<(unsigned)((n + 255) / 256), 256, 0, stream>>>(W, Wt, N, K);
  };
  auto gemm = [&](const bf16* X, const bf16* Wt, float* Y, int m, int n, int k) {
    dim3 g((n + GBN - 1) / GBN, (m + GBM - 1) / GBM);
    gemm_xwt_kernel<<<g, 256, 0, stream>>>(X, Wt, Y, m, n, k);
  };

  // h = hidden_states (don't mutate inputs)
  hipMemcpyAsync(hbuf, hid, M * HH * sizeof(float), hipMemcpyDeviceToDevice, stream);

  const size_t NE = M * HH;      // 4.19M elements
  const size_t NF = M * FF;      // 11.2M elements

  for (int l = 0; l < NLAYER; l++) {
    // per-layer bf16 transposed weights
    castT(Wq  + (size_t)l * HxH, WqT,  HH, HH);
    castT(Wk  + (size_t)l * HxH, WkT,  HH, HH);
    castT(Wv  + (size_t)l * HxH, WvT,  HH, HH);
    castT(Wo  + (size_t)l * HxH, WoT,  HH, HH);
    castT(Wg  + (size_t)l * FxH, WgT,  FF, HH);
    castT(Wvf + (size_t)l * FxH, WvfT, FF, HH);
    castT(Wof + (size_t)l * FxH, WofT, HH, FF);

    // --- attention block ---
    rmsnorm_bf16_kernel<<<(unsigned)M, 256, 0, stream>>>(hbuf, nPreA + l * HH, xb);
    gemm(xb, WqT, big1, (int)M, HH, HH);
    gemm(xb, WkT, big2, (int)M, HH, HH);
    gemm(xb, WvT, sml,  (int)M, HH, HH);
    rope_q_kernel<<<(unsigned)((NE + 255) / 256), 256, 0, stream>>>(big1, qb);
    rope_kT_kernel<<<(unsigned)((NE + 255) / 256), 256, 0, stream>>>(big2, kt);
    cast_bf16_kernel<<<(unsigned)((NE + 255) / 256), 256, 0, stream>>>(sml, vb, NE);
    attn_kernel<<<dim3(BB * NHEAD, SS / 64), 128, 0, stream>>>(qb, kt, vb, pad, attb);
    gemm(attb, WoT, sml, (int)M, HH, HH);
    resid_post_kernel<<<(unsigned)M, 256, 0, stream>>>(hbuf, sml, nPostA + l * HH, pad, 1, 0);

    // --- FFN block ---
    rmsnorm_bf16_kernel<<<(unsigned)M, 256, 0, stream>>>(hbuf, nPreF + l * HH, xb);
    gemm(xb, WgT,  big1, (int)M, FF, HH);
    gemm(xb, WvfT, big2, (int)M, FF, HH);
    gelu_gate_kernel<<<(unsigned)((NF + 255) / 256), 256, 0, stream>>>(big1, big2, gate, NF);
    gemm(gate, WofT, sml, (int)M, HH, FF);
    resid_post_kernel<<<(unsigned)M, 256, 0, stream>>>(hbuf, sml, nPostF + l * HH, pad, 0, 1);
  }

  final_rms_kernel<<<(unsigned)M, 256, 0, stream>>>(hbuf, nFinal, (float*)d_out);
}
